// MyViT_44495861186632
// MI455X (gfx1250) — matrix operand order repaired and checked by reading, not compiled, not run on hardware
//
#include <hip/hip_runtime.h>
#include <hip/hip_bf16.h>

typedef __attribute__((ext_vector_type(16))) _Float16 v16h;
typedef __attribute__((ext_vector_type(8)))  float    v8f;
typedef __attribute__((ext_vector_type(4)))  unsigned u32x4;
typedef __attribute__((ext_vector_type(8)))  unsigned u32x8;

#define DEV __device__ __forceinline__

// ---------------- WMMA fragment helpers (wave32, 16x16x32 f16) ----------------
// A: 16x32 (MxK). Lanes 0-15: M=lane, elems 0..15 -> K=kk+0..15
//                 Lanes16-31: M=lane-16, elems 0..15 -> K=kk+16..31.
DEV v16h load_a16x32(const _Float16* a, long ld, int m0, int mMax, int kk) {
  int lane = threadIdx.x & 31;
  int m = m0 + (lane & 15); if (m > mMax) m = mMax;
  int kb = (lane & 16) ? 16 : 0;
  const _Float16* p = a + (long)m * ld + kk + kb;
  v16h r;
#pragma unroll
  for (int i = 0; i < 16; ++i) r[i] = p[i];
  return r;
}

// B: 32x16 (KxN), element(k,n) = b[n*ld + k]   (W[n][k] row-major)
DEV v16h load_b_nk(const _Float16* b, long ld, int n0, int nMax, int kk) {
  int lane = threadIdx.x & 31;
  int n = n0 + (lane & 15); if (n > nMax) n = nMax;
  int kb = (lane & 16) ? 16 : 0;
  const _Float16* p = b + (long)n * ld + kk + kb;
  v16h r;
#pragma unroll
  for (int i = 0; i < 16; ++i) r[i] = p[i];
  return r;
}

// B: 32x16 (KxN), element(k,n) = b[k*ld + n]   (row-major in K)
DEV v16h load_b_kn(const _Float16* b, long ld, int n0, int nMax, int kk) {
  int lane = threadIdx.x & 31;
  int n = n0 + (lane & 15); if (n > nMax) n = nMax;
  int kb = (lane & 16) ? 16 : 0;
  const _Float16* p = b + (long)(kk + kb) * ld + n;
  v16h r;
#pragma unroll
  for (int i = 0; i < 16; ++i) r[i] = p[(long)i * ld];
  return r;
}

DEV v8f wmma_f16(v16h a, v16h b, v8f c) {
  return __builtin_amdgcn_wmma_f32_16x16x32_f16(false, a, false, b, (short)0, c, false, false);
}

// ---------------- Tensor Data Mover helpers ----------------
DEV void set_bits(unsigned* w, int lo, int width, unsigned long long v) {
  if (width < 64) v &= ((1ull << width) - 1ull);
  int dw = lo >> 5, sh = lo & 31;
  w[dw] |= (unsigned)(v << sh);
  if (sh + width > 32) w[dw + 1] |= (unsigned)(v >> (32 - sh));
  if (sh + width > 64) w[dw + 2] |= (unsigned)(v >> (64 - sh));
}

// 2-D f16 tile load: global (tensor_d0 x tensor_d1, stride0 elems/row) -> LDS
DEV void tdm_load_2d_f16(unsigned lds_addr, const void* gaddr,
                         unsigned tensor_d0, unsigned tensor_d1,
                         unsigned tile_d0, unsigned tile_d1, unsigned stride0) {
  unsigned g0a[4] = {0, 0, 0, 0};
  unsigned g1a[8] = {0, 0, 0, 0, 0, 0, 0, 0};
  unsigned long long ga = (unsigned long long)(size_t)gaddr;
  g0a[0] = 1u;                                   // count=1, user descriptor
  g0a[1] = lds_addr;                             // LDS byte address
  g0a[2] = (unsigned)(ga & 0xffffffffu);         // global_addr[31:0]
  g0a[3] = (unsigned)((ga >> 32) & 0x01ffffffu)  // global_addr[56:32]
         | (2u << 30);                           // type = 2 ("image")
  set_bits(g1a, 16, 2, 1);                       // data_size = 2 bytes
  set_bits(g1a, 48, 32, tensor_d0);              // tensor_dim0
  set_bits(g1a, 80, 32, tensor_d1);              // tensor_dim1
  set_bits(g1a, 112, 16, tile_d0);               // tile_dim0
  set_bits(g1a, 128, 16, tile_d1);               // tile_dim1
  set_bits(g1a, 160, 48, stride0);               // tensor_dim0_stride
  u32x4 g0; u32x8 g1;
#pragma unroll
  for (int i = 0; i < 4; ++i) g0[i] = g0a[i];
#pragma unroll
  for (int i = 0; i < 8; ++i) g1[i] = g1a[i];
  asm volatile("tensor_load_to_lds %0, %1" :: "s"(g0), "s"(g1) : "memory");
}

// ---------------- Elementwise / setup kernels ----------------
__global__ void k_cvt16(const float* __restrict__ s, _Float16* __restrict__ d, int n) {
  for (int i = blockIdx.x * blockDim.x + threadIdx.x; i < n; i += gridDim.x * blockDim.x)
    d[i] = (_Float16)s[i];
}

__global__ void k_pos(float* __restrict__ pos) {
  int idx = blockIdx.x * blockDim.x + threadIdx.x;
  if (idx >= 197 * 768) return;
  int j = idx % 768, i = idx / 768;
  float jeff = (float)(j & ~1);
  float ang = (float)i * __expf(jeff * (-9.210340371976184f / 768.0f)); // i / 10000^(jeff/768)
  pos[idx] = (j & 1) ? __cosf(ang) : __sinf(ang);
}

// conv 5x5 stride4 pad1 + bias + BN + ReLU, written directly in patch-major (n,196,800) f16 layout
__global__ void k_conv(const float* __restrict__ img, const float* __restrict__ cw,
                       const float* __restrict__ cb, const float* __restrict__ bg,
                       const float* __restrict__ bb, const float* __restrict__ bm,
                       const float* __restrict__ bnv, _Float16* __restrict__ patches) {
  int idx = blockIdx.x * blockDim.x + threadIdx.x;
  const int TOT = 32 * 32 * 70 * 70;
  if (idx >= TOT) return;
  int x = idx % 70;
  int y = (idx / 70) % 70;
  int c = (idx / 4900) % 32;
  int n = idx / (4900 * 32);
  float acc = 0.f;
  const float* wbase = cw + c * 75;
  const float* ibase = img + (long)n * 3 * 280 * 280;
  int iy0 = y * 4 - 1, ix0 = x * 4 - 1;
#pragma unroll
  for (int ic = 0; ic < 3; ++ic) {
    const float* ip = ibase + ic * 280 * 280;
    const float* wp = wbase + ic * 25;
#pragma unroll
    for (int ky = 0; ky < 5; ++ky) {
      int iy = iy0 + ky;
      if (iy < 0 || iy >= 280) continue;
#pragma unroll
      for (int kx = 0; kx < 5; ++kx) {
        int ix = ix0 + kx;
        if (ix < 0 || ix >= 280) continue;
        acc += ip[iy * 280 + ix] * wp[ky * 5 + kx];
      }
    }
  }
  float scale = bg[c] * rsqrtf(bnv[c] + 1e-5f);
  float val = (acc + cb[c] - bm[c]) * scale + bb[c];
  val = fmaxf(val, 0.f);
  int ph = y / 5, i5 = y % 5, pw = x / 5, j5 = x % 5;
  long po = ((long)n * 196 + ph * 14 + pw) * 800 + c * 25 + i5 * 5 + j5;
  patches[po] = (_Float16)val;
}

__global__ void k_cls(const float* __restrict__ cls_tok, const float* __restrict__ pos,
                      float* __restrict__ tokens) {
  int idx = blockIdx.x * blockDim.x + threadIdx.x;
  if (idx >= 32 * 768) return;
  int e = idx % 768, n = idx / 768;
  tokens[(long)n * 197 * 768 + e] = cls_tok[e] + pos[e];
}

// ---------------- GEMM kernels ----------------
// tokens[n, p+1, :] = patches @ map_w^T + map_b + pos[p+1]
// one wave computes a 16x64 C block (4 N-tiles) => A fragment reused 4x
__global__ __launch_bounds__(32) void k_patch_gemm(const _Float16* __restrict__ A,
                                                   const _Float16* __restrict__ W,
                                                   const float* __restrict__ bias,
                                                   const float* __restrict__ pos,
                                                   float* __restrict__ tokens) {
  int bid = blockIdx.x;            // 392 M-tiles x 12 N-groups
  int ng = bid % 12, mt = bid / 12;
  int m0 = mt * 16, n0 = ng * 64;
  v8f c[4] = {v8f{}, v8f{}, v8f{}, v8f{}};
  for (int kk = 0; kk < 800; kk += 32) {
    v16h a = load_a16x32(A, 800, m0, 6271, kk);
#pragma unroll
    for (int et = 0; et < 4; ++et) {
      v16h b = load_b_nk(W, 800, n0 + et * 16, 767, kk);
      c[et] = wmma_f16(a, b, c[et]);
    }
  }
  int lane = threadIdx.x & 31;
  int rb = (lane & 16) ? 8 : 0;
#pragma unroll
  for (int et = 0; et < 4; ++et) {
    int col = n0 + et * 16 + (lane & 15);
#pragma unroll
    for (int r = 0; r < 8; ++r) {
      int m = m0 + rb + r;
      int n = m / 196, p = m % 196;
      tokens[((long)n * 197 + p + 1) * 768 + col] = c[et][r] + bias[col] + pos[(p + 1) * 768 + col];
    }
  }
}

__global__ void k_ln(const float* __restrict__ tokens, const float* __restrict__ g,
                     const float* __restrict__ b, _Float16* __restrict__ h) {
  __shared__ float s1[256], s2[256];
  int row = blockIdx.x;            // 32*197 rows
  const float* x = tokens + (long)row * 768;
  float sum = 0.f, sq = 0.f;
  for (int i = threadIdx.x; i < 768; i += 256) { float v = x[i]; sum += v; sq += v * v; }
  s1[threadIdx.x] = sum; s2[threadIdx.x] = sq;
  __syncthreads();
  for (int s = 128; s > 0; s >>= 1) {
    if ((int)threadIdx.x < s) { s1[threadIdx.x] += s1[threadIdx.x + s]; s2[threadIdx.x] += s2[threadIdx.x + s]; }
    __syncthreads();
  }
  float mu  = s1[0] * (1.f / 768.f);
  float var = s2[0] * (1.f / 768.f) - mu * mu;
  float inv = rsqrtf(var + 1e-5f);
  _Float16* o = h + (long)row * 768;
  for (int i = threadIdx.x; i < 768; i += 256)
    o[i] = (_Float16)((x[i] - mu) * inv * g[i] + b[i]);
}

// per-head 64x64 projection: one wave = 16 rows x full 64 cols (4 wmma N-tiles, A reused)
__global__ __launch_bounds__(32) void k_qkv(const _Float16* __restrict__ h,
                                            const _Float16* __restrict__ w,
                                            const float* __restrict__ bias,
                                            _Float16* __restrict__ out) {
  int bid = blockIdx.x;            // 32*12*13
  int st = bid % 13;
  int head = (bid / 13) % 12;
  int n = bid / 156;
  int m0 = st * 16;
  const _Float16* Ab = h + (long)n * 197 * 768 + head * 64;  // row stride 768
  const _Float16* Wb = w + head * 4096;                       // (64,64) row-major
  v8f c[4] = {v8f{}, v8f{}, v8f{}, v8f{}};
  for (int kk = 0; kk < 64; kk += 32) {
    v16h a = load_a16x32(Ab, 768, m0, 196, kk);
#pragma unroll
    for (int et = 0; et < 4; ++et) {
      v16h b = load_b_nk(Wb, 64, et * 16, 63, kk);
      c[et] = wmma_f16(a, b, c[et]);
    }
  }
  int lane = threadIdx.x & 31;
  int rb = (lane & 16) ? 8 : 0;
  _Float16* ob = out + (long)(n * 12 + head) * 197 * 64;
#pragma unroll
  for (int et = 0; et < 4; ++et) {
    int e = et * 16 + (lane & 15);
    float bval = bias[head * 64 + e];
#pragma unroll
    for (int r = 0; r < 8; ++r) {
      int s = m0 + rb + r;
      if (s < 197) ob[(long)s * 64 + e] = (_Float16)(c[et][r] + bval);
    }
  }
}

// scores: one 256-thread block per (n,head); K (197x64 f16) staged to LDS via TDM,
// 8 waves cover the 13x13 score tiles. attn[n,h,s,t] = (q.k)/8, f16, stride 224.
__global__ __launch_bounds__(256) void k_scores(const _Float16* __restrict__ q,
                                                const _Float16* __restrict__ kmat,
                                                _Float16* __restrict__ attn) {
  __shared__ _Float16 ksh[197 * 64];
  int nh = blockIdx.x;             // n*12+head, 384 blocks
  long base = (long)nh * 197 * 64;
  if (threadIdx.x < 32) {
    tdm_load_2d_f16((unsigned)(size_t)&ksh[0], kmat + base, 64, 197, 64, 197, 64);
    __builtin_amdgcn_s_wait_tensorcnt(0);
  }
  __syncthreads();
  const _Float16* Q = q + base;
  _Float16* ab = attn + (long)nh * 197 * 224;
  int wid = threadIdx.x >> 5;
  int lane = threadIdx.x & 31;
  int rb = (lane & 16) ? 8 : 0;
  for (int tile = wid; tile < 169; tile += 8) {
    int st = tile / 13, tt = tile % 13;
    int m0 = st * 16, n0 = tt * 16;
    v8f c = {};
    for (int kk = 0; kk < 64; kk += 32) {
      v16h a = load_a16x32(Q, 64, m0, 196, kk);
      v16h b = load_b_nk(ksh, 64, n0, 196, kk);  // B[e][t] = K[t*64+e] from LDS
      c = wmma_f16(a, b, c);
    }
    int t = n0 + (lane & 15);
#pragma unroll
    for (int r = 0; r < 8; ++r) {
      int s = m0 + rb + r;
      if (s < 197) ab[(long)s * 224 + t] = (_Float16)(c[r] * 0.125f);
    }
  }
}

// in-place row softmax over t<197, zero-pad t in [197,224)
__global__ void k_attn_softmax(_Float16* __restrict__ attn) {
  __shared__ float red[256];
  int row = blockIdx.x;            // 32*12*197 rows
  _Float16* p = attn + (long)row * 224;
  int t = threadIdx.x;             // 256 threads
  float v = (t < 197) ? (float)p[t] : -1e30f;
  red[t] = v; __syncthreads();
  for (int s = 128; s > 0; s >>= 1) { if (t < s) red[t] = fmaxf(red[t], red[t + s]); __syncthreads(); }
  float mx = red[0]; __syncthreads();
  float e = (t < 197) ? __expf(v - mx) : 0.f;
  red[t] = e; __syncthreads();
  for (int s = 128; s > 0; s >>= 1) { if (t < s) red[t] += red[t + s]; __syncthreads(); }
  float inv = 1.f / red[0];
  if (t < 224) p[t] = (t < 197) ? (_Float16)(e * inv) : (_Float16)0.f;
}

// attn @ V: one block per (n,head); V (197x64) staged to LDS via TDM, pad rows zeroed,
// 8 waves cover 13x4 output tiles; residual-add into tokens.
__global__ __launch_bounds__(256) void k_attn_out(const _Float16* __restrict__ attn,
                                                  const _Float16* __restrict__ v,
                                                  float* __restrict__ tokens) {
  __shared__ _Float16 vsh[224 * 64];
  int nh = blockIdx.x;             // 384 blocks
  int n = nh / 12, head = nh % 12;
  for (int i = threadIdx.x; i < 27 * 64; i += 256)   // zero K-pad rows 197..223
    vsh[197 * 64 + i] = (_Float16)0.f;
  if (threadIdx.x < 32) {
    tdm_load_2d_f16((unsigned)(size_t)&vsh[0], v + (long)nh * 197 * 64, 64, 197, 64, 197, 64);
    __builtin_amdgcn_s_wait_tensorcnt(0);
  }
  __syncthreads();
  const _Float16* A = attn + (long)nh * 197 * 224;   // ld 224
  float* tbase = tokens + (long)n * 197 * 768 + head * 64;
  int wid = threadIdx.x >> 5;
  int lane = threadIdx.x & 31;
  int rb = (lane & 16) ? 8 : 0;
  for (int tile = wid; tile < 52; tile += 8) {
    int st = tile >> 2, et = tile & 3;
    int m0 = st * 16, n0 = et * 16;
    v8f c = {};
    for (int kk = 0; kk < 224; kk += 32) {
      v16h a = load_a16x32(A, 224, m0, 196, kk);
      v16h b = load_b_kn(vsh, 64, n0, 63, kk);       // V[t][e] from LDS
      c = wmma_f16(a, b, c);
    }
    int e = n0 + (lane & 15);
#pragma unroll
    for (int r = 0; r < 8; ++r) {
      int s = m0 + rb + r;
      if (s < 197) tbase[(long)s * 768 + e] += c[r];
    }
  }
}

__global__ void k_cls16(const float* __restrict__ tokens, _Float16* __restrict__ cls16) {
  int idx = blockIdx.x * blockDim.x + threadIdx.x;
  if (idx >= 32 * 768) return;
  int e = idx % 768, n = idx / 768;
  cls16[idx] = (_Float16)tokens[(long)n * 197 * 768 + e];
}

__global__ __launch_bounds__(32) void k_head(const _Float16* __restrict__ cls16,
                                             const _Float16* __restrict__ W,
                                             const float* __restrict__ bias,
                                             float* __restrict__ logits) {
  int bid = blockIdx.x;            // 2 x 63 tiles
  int nt = bid % 63, mt = bid / 63;
  int m0 = mt * 16, n0 = nt * 16;
  v8f c = {};
  for (int kk = 0; kk < 768; kk += 32) {
    v16h a = load_a16x32(cls16, 768, m0, 31, kk);
    v16h b = load_b_nk(W, 768, n0, 999, kk);
    c = wmma_f16(a, b, c);
  }
  int lane = threadIdx.x & 31;
  int col = n0 + (lane & 15);
  int rb = (lane & 16) ? 8 : 0;
  if (col < 1000) {
#pragma unroll
    for (int r = 0; r < 8; ++r) {
      int m = m0 + rb + r;
      logits[(long)m * 1000 + col] = c[r] + bias[col];
    }
  }
}

__global__ void k_out_softmax(const float* __restrict__ logits, float* __restrict__ out) {
  __shared__ float red[256];
  int n = blockIdx.x;
  const float* x = logits + (long)n * 1000;
  float mx = -1e30f;
  for (int i = threadIdx.x; i < 1000; i += 256) mx = fmaxf(mx, x[i]);
  red[threadIdx.x] = mx; __syncthreads();
  for (int s = 128; s > 0; s >>= 1) { if ((int)threadIdx.x < s) red[threadIdx.x] = fmaxf(red[threadIdx.x], red[threadIdx.x + s]); __syncthreads(); }
  mx = red[0]; __syncthreads();
  float sum = 0.f;
  for (int i = threadIdx.x; i < 1000; i += 256) sum += __expf(x[i] - mx);
  red[threadIdx.x] = sum; __syncthreads();
  for (int s = 128; s > 0; s >>= 1) { if ((int)threadIdx.x < s) red[threadIdx.x] += red[threadIdx.x + s]; __syncthreads(); }
  float inv = 1.f / red[0];
  for (int i = threadIdx.x; i < 1000; i += 256) out[(long)n * 1000 + i] = __expf(x[i] - mx) * inv;
}

// ---------------- launch ----------------
extern "C" void kernel_launch(void* const* d_in, const int* in_sizes, int n_in,
                              void* d_out, int out_size, void* d_ws, size_t ws_size,
                              hipStream_t stream) {
  const float* images  = (const float*)d_in[0];
  const float* conv_w  = (const float*)d_in[1];
  const float* conv_b  = (const float*)d_in[2];
  const float* bn_g    = (const float*)d_in[3];
  const float* bn_b    = (const float*)d_in[4];
  const float* bn_m    = (const float*)d_in[5];
  const float* bn_v    = (const float*)d_in[6];
  const float* map_w   = (const float*)d_in[7];
  const float* map_b   = (const float*)d_in[8];
  const float* cls_tok = (const float*)d_in[9];
  const float* ln_g    = (const float*)d_in[10];
  const float* ln_b    = (const float*)d_in[11];
  const float* wq      = (const float*)d_in[12];
  const float* bq      = (const float*)d_in[13];
  const float* wk      = (const float*)d_in[14];
  const float* bk      = (const float*)d_in[15];
  const float* wv      = (const float*)d_in[16];
  const float* bvv     = (const float*)d_in[17];
  const float* mlp_w   = (const float*)d_in[18];
  const float* mlp_b   = (const float*)d_in[19];
  float* out = (float*)d_out;
  (void)in_sizes; (void)n_in; (void)out_size; (void)ws_size;

  char* ws = (char*)d_ws;
  size_t off = 0;
  auto alloc = [&](size_t bytes) -> void* {
    void* p = ws + off;
    off = (off + bytes + 255) & ~(size_t)255;
    return p;
  };

  _Float16* patches = (_Float16*)alloc((size_t)32 * 196 * 800 * 2);
  _Float16* mapw16  = (_Float16*)alloc((size_t)768 * 800 * 2);
  _Float16* wq16    = (_Float16*)alloc((size_t)12 * 12 * 64 * 64 * 2);
  _Float16* wk16    = (_Float16*)alloc((size_t)12 * 12 * 64 * 64 * 2);
  _Float16* wv16    = (_Float16*)alloc((size_t)12 * 12 * 64 * 64 * 2);
  _Float16* mlpw16  = (_Float16*)alloc((size_t)1000 * 768 * 2);
  float*    pos     = (float*)alloc((size_t)197 * 768 * 4);
  float*    tokens  = (float*)alloc((size_t)32 * 197 * 768 * 4);
  _Float16* h16     = (_Float16*)alloc((size_t)32 * 197 * 768 * 2);
  _Float16* q16     = (_Float16*)alloc((size_t)32 * 12 * 197 * 64 * 2);
  _Float16* k16     = (_Float16*)alloc((size_t)32 * 12 * 197 * 64 * 2);
  _Float16* v16b    = (_Float16*)alloc((size_t)32 * 12 * 197 * 64 * 2);
  _Float16* attn16  = (_Float16*)alloc((size_t)32 * 12 * 197 * 224 * 2);
  _Float16* cls16   = (_Float16*)alloc((size_t)32 * 768 * 2);
  float*    logits  = (float*)alloc((size_t)32 * 1000 * 4);

  // f16 weight copies (stateless: redone every call)
  k_cvt16<<<(768 * 800 + 255) / 256, 256, 0, stream>>>(map_w, mapw16, 768 * 800);
  k_cvt16<<<(589824 + 255) / 256, 256, 0, stream>>>(wq, wq16, 589824);
  k_cvt16<<<(589824 + 255) / 256, 256, 0, stream>>>(wk, wk16, 589824);
  k_cvt16<<<(589824 + 255) / 256, 256, 0, stream>>>(wv, wv16, 589824);
  k_cvt16<<<(768000 + 255) / 256, 256, 0, stream>>>(mlp_w, mlpw16, 768000);

  k_pos<<<(197 * 768 + 255) / 256, 256, 0, stream>>>(pos);

  k_conv<<<(32 * 32 * 70 * 70 + 255) / 256, 256, 0, stream>>>(
      images, conv_w, conv_b, bn_g, bn_b, bn_m, bn_v, patches);
  k_cls<<<(32 * 768 + 255) / 256, 256, 0, stream>>>(cls_tok, pos, tokens);
  k_patch_gemm<<<392 * 12, 32, 0, stream>>>(patches, mapw16, map_b, pos, tokens);

  for (int l = 0; l < 12; ++l) {
    k_ln<<<32 * 197, 256, 0, stream>>>(tokens, ln_g + l * 768, ln_b + l * 768, h16);
    k_qkv<<<32 * 12 * 13, 32, 0, stream>>>(h16, wq16 + (size_t)l * 12 * 4096, bq + l * 12 * 64, q16);
    k_qkv<<<32 * 12 * 13, 32, 0, stream>>>(h16, wk16 + (size_t)l * 12 * 4096, bk + l * 12 * 64, k16);
    k_qkv<<<32 * 12 * 13, 32, 0, stream>>>(h16, wv16 + (size_t)l * 12 * 4096, bvv + l * 12 * 64, v16b);
    k_scores<<<384, 256, 0, stream>>>(q16, k16, attn16);
    k_attn_softmax<<<32 * 12 * 197, 256, 0, stream>>>(attn16);
    k_attn_out<<<384, 256, 0, stream>>>(attn16, v16b, tokens);
  }

  k_cls16<<<(32 * 768 + 255) / 256, 256, 0, stream>>>(tokens, cls16);
  k_head<<<2 * 63, 32, 0, stream>>>(cls16, mlpw16, mlp_b, logits);
  k_out_softmax<<<32, 256, 0, stream>>>(logits, out);
}
